// Net_43422119363222
// MI455X (gfx1250) — compile-verified, matrix operand order checked by reading, not run on hardware
//
#include <hip/hip_runtime.h>
#include <hip/hip_bf16.h>

// ---------------------------------------------------------------------------
// MI455X (gfx1250) implementation.
//  - All matmuls on v_wmma_f32_16x16x32_bf16 (wave32), f32 accumulation.
//  - K/Q in bf16 row-major; V in bf16 TRANSPOSED (Vt[d][n]) so the attention
//    P^T@V B-fragments are contiguous 128-bit global loads.
//  - Banded softmax-over-queries attention fused in one kernel; probability
//    panel kept transposed in LDS so A-fragments are contiguous ds_load_b128.
// Workspace layout (needs >= 72 MB):
//   [0,   8MB) Kb  bf16 4096x1024
//   [8,  16MB) Qb  bf16 4096x1024 (already scaled by 0.06)
//   [16, 24MB) Vt  bf16 1024x4096  (transposed V)
//   [24, 40MB) yatt f32 4096x1024
//   [40, 56MB) y2   f32 4096x1024
//   [56, 72MB) t1   f32 4096x1024
// ---------------------------------------------------------------------------

typedef __attribute__((ext_vector_type(16))) __bf16 v16bf;
typedef __attribute__((ext_vector_type(8)))  __bf16 v8bf;
typedef __attribute__((ext_vector_type(8)))  float  v8f;

#define NSEQ 4096
#define DMODEL 1024
#define APERT 256
#define LN_EPS 1e-3f

// k index inside a 16-bit A/B fragment (ISA 7.12.2: 16-bit A 16x32 layout).
// For fixed half, elements 0..7 sit at k = half*8 + 0..7 and elements 8..15
// at k = 16 + half*8 + 0..7  -> two contiguous 8-element runs.
__device__ __forceinline__ int kmap(int t, int half) {
  int v = t >> 1, e = t & 1;
  return ((v < 4) ? 0 : 16) + half * 8 + (v & 3) * 2 + e;
}

__device__ __forceinline__ v8f wmma_bf16(v16bf a, v16bf b, v8f c) {
  return __builtin_amdgcn_wmma_f32_16x16x32_bf16(
      false, a, false, b, (short)0, c, false, false);
}

// Build a 16x32-slice fragment from two contiguous 8xbf16 runs.
__device__ __forceinline__ v16bf frag_from_runs(const __bf16* run0,
                                                const __bf16* run1) {
  const v8bf r0 = *(const v8bf*)run0;
  const v8bf r1 = *(const v8bf*)run1;
  v16bf f;
  #pragma unroll
  for (int t = 0; t < 8; ++t) { f[t] = r0[t]; f[8 + t] = r1[t]; }
  return f;
}

// ---------------------------------------------------------------------------
// GEMM: C(bf16) = scale * (A(f32)[N,1024] @ B(f32)[1024,1024])
// Block tile 128x64, 8 waves (4 in M x 2 in N), each wave 32x32 (2x2 WMMA).
// TR=false: C row-major [N,1024]. TR=true: C transposed [1024, ldc=NSEQ].
// ---------------------------------------------------------------------------
#define BM 128
#define BN 64
#define BK 32

template <bool TR>
__global__ __launch_bounds__(256)
void gemm_bf16out(const float* __restrict__ A, const float* __restrict__ B,
                  __bf16* __restrict__ C, float scale) {
  __shared__ __bf16 As[BM][BK];   // 8 KB
  __shared__ __bf16 Bs[BN][BK];   // 4 KB (k-major per column -> frag reads)
  const int m0 = blockIdx.x * BM;
  const int n0 = blockIdx.y * BN;
  const int tid = threadIdx.x;
  const int lane = tid & 31, wid = tid >> 5;
  const int hrow = lane & 15, half = lane >> 4;
  const int wy = wid & 3, wx = wid >> 2;

  v8f acc[2][2];
  #pragma unroll
  for (int i = 0; i < 2; ++i)
    #pragma unroll
    for (int j = 0; j < 2; ++j) acc[i][j] = (v8f){};

  for (int k0 = 0; k0 < DMODEL; k0 += BK) {
    #pragma unroll
    for (int it = 0; it < (BM * BK) / 256; ++it) {
      int idx = tid + it * 256;
      int r = idx >> 5, c = idx & 31;
      As[r][c] = (__bf16)A[(size_t)(m0 + r) * DMODEL + k0 + c];
    }
    #pragma unroll
    for (int it = 0; it < (BN * BK) / 256; ++it) {
      int idx = tid + it * 256;
      int n = idx >> 5, c = idx & 31;
      Bs[n][c] = (__bf16)B[(size_t)(k0 + c) * DMODEL + n0 + n];
    }
    __syncthreads();

    v16bf af[2], bf[2];
    #pragma unroll
    for (int ms = 0; ms < 2; ++ms) {
      const __bf16* r = &As[wy * 32 + ms * 16 + hrow][half * 8];
      af[ms] = frag_from_runs(r, r + 16);
    }
    #pragma unroll
    for (int ns = 0; ns < 2; ++ns) {
      const __bf16* r = &Bs[wx * 32 + ns * 16 + hrow][half * 8];
      bf[ns] = frag_from_runs(r, r + 16);
    }
    #pragma unroll
    for (int ms = 0; ms < 2; ++ms)
      #pragma unroll
      for (int ns = 0; ns < 2; ++ns)
        acc[ms][ns] = wmma_bf16(af[ms], bf[ns], acc[ms][ns]);
    __syncthreads();
  }

  #pragma unroll
  for (int ms = 0; ms < 2; ++ms)
    #pragma unroll
    for (int ns = 0; ns < 2; ++ns) {
      int col = n0 + wx * 32 + ns * 16 + hrow;
      #pragma unroll
      for (int r = 0; r < 8; ++r) {
        int row = m0 + wy * 32 + ms * 16 + r + 8 * half;
        __bf16 v = (__bf16)(scale * acc[ms][ns][r]);
        if (TR) C[(size_t)col * NSEQ + row] = v;       // transposed store
        else    C[(size_t)row * DMODEL + col] = v;
      }
    }
}

// ---------------------------------------------------------------------------
// GEMM: C(f32) = A(f32) @ B(f32) with epilogue
//   MODE 0: + residual R      MODE 1: relu(+ bias)
// ---------------------------------------------------------------------------
template <int MODE>
__global__ __launch_bounds__(256)
void gemm_f32out(const float* __restrict__ A, const float* __restrict__ B,
                 float* __restrict__ C, const float* __restrict__ R,
                 const float* __restrict__ bias) {
  __shared__ __bf16 As[BM][BK];
  __shared__ __bf16 Bs[BN][BK];
  const int m0 = blockIdx.x * BM;
  const int n0 = blockIdx.y * BN;
  const int tid = threadIdx.x;
  const int lane = tid & 31, wid = tid >> 5;
  const int hrow = lane & 15, half = lane >> 4;
  const int wy = wid & 3, wx = wid >> 2;

  v8f acc[2][2];
  #pragma unroll
  for (int i = 0; i < 2; ++i)
    #pragma unroll
    for (int j = 0; j < 2; ++j) acc[i][j] = (v8f){};

  for (int k0 = 0; k0 < DMODEL; k0 += BK) {
    #pragma unroll
    for (int it = 0; it < (BM * BK) / 256; ++it) {
      int idx = tid + it * 256;
      int r = idx >> 5, c = idx & 31;
      As[r][c] = (__bf16)A[(size_t)(m0 + r) * DMODEL + k0 + c];
    }
    #pragma unroll
    for (int it = 0; it < (BN * BK) / 256; ++it) {
      int idx = tid + it * 256;
      int n = idx >> 5, c = idx & 31;
      Bs[n][c] = (__bf16)B[(size_t)(k0 + c) * DMODEL + n0 + n];
    }
    __syncthreads();

    v16bf af[2], bf[2];
    #pragma unroll
    for (int ms = 0; ms < 2; ++ms) {
      const __bf16* r = &As[wy * 32 + ms * 16 + hrow][half * 8];
      af[ms] = frag_from_runs(r, r + 16);
    }
    #pragma unroll
    for (int ns = 0; ns < 2; ++ns) {
      const __bf16* r = &Bs[wx * 32 + ns * 16 + hrow][half * 8];
      bf[ns] = frag_from_runs(r, r + 16);
    }
    #pragma unroll
    for (int ms = 0; ms < 2; ++ms)
      #pragma unroll
      for (int ns = 0; ns < 2; ++ns)
        acc[ms][ns] = wmma_bf16(af[ms], bf[ns], acc[ms][ns]);
    __syncthreads();
  }

  #pragma unroll
  for (int ms = 0; ms < 2; ++ms)
    #pragma unroll
    for (int ns = 0; ns < 2; ++ns) {
      int col = n0 + wx * 32 + ns * 16 + hrow;
      #pragma unroll
      for (int r = 0; r < 8; ++r) {
        int row = m0 + wy * 32 + ms * 16 + r + 8 * half;
        float v = acc[ms][ns][r];
        if (MODE == 0) v += R[(size_t)row * DMODEL + col];
        else           v = fmaxf(v + bias[col], 0.0f);
        C[(size_t)row * DMODEL + col] = v;
      }
    }
}

// ---------------------------------------------------------------------------
// Banded attention with softmax over queries (axis 0).
// Block = 128 threads (4 waves), handles 16 output rows j.
//   y[j,:] = sum_{i in [j-256,j+256]} softmax_i(Q[i].K[j]) * V[i,:]
// Band panel padded to 544 rows (34 tiles of 16); pad rows are band-masked.
// ---------------------------------------------------------------------------
#define JT 16
#define ITILES 34
#define ILEN (ITILES * 16)   // 544 (multiple of 32 for the K=32 reduction)

__global__ __launch_bounds__(128)
void attn_banded(const __bf16* __restrict__ Qb, const __bf16* __restrict__ Kb,
                 const __bf16* __restrict__ Vt, float* __restrict__ Y) {
  __shared__ float Pt[JT][ILEN];  // transposed prob panel, ~34 KB
  __shared__ float red[JT][8];

  const int j0 = blockIdx.x * JT;
  const int i0 = j0 - APERT;                  // may be negative
  const int tid = threadIdx.x;
  const int lane = tid & 31, wid = tid >> 5;  // 4 waves
  const int hrow = lane & 15, half = lane >> 4;

  // ---- phase 1: S[i_local, j] = Q[i].K[j], band-masked --------------------
  for (int t = wid; t < ITILES; t += 4) {
    const int ib = i0 + t * 16;
    int qi = ib + hrow;
    qi = (qi < 0) ? 0 : (qi >= NSEQ ? NSEQ - 1 : qi);  // clamp; masked below
    const __bf16* qrow = Qb + (size_t)qi * DMODEL;
    const __bf16* krow = Kb + (size_t)(j0 + hrow) * DMODEL;
    v8f acc = (v8f){};
    for (int kk = 0; kk < DMODEL; kk += 32) {
      v16bf a = frag_from_runs(qrow + kk + half * 8, qrow + kk + 16 + half * 8);
      v16bf b = frag_from_runs(krow + kk + half * 8, krow + kk + 16 + half * 8);
      acc = wmma_bf16(a, b, acc);
    }
    #pragma unroll
    for (int r = 0; r < 8; ++r) {
      const int il = t * 16 + r + 8 * half;     // i_local
      const int gi = i0 + il;
      const int gj = j0 + hrow;
      float v = acc[r];
      const int dd = gi - gj;
      if (gi < 0 || gi >= NSEQ || dd > APERT || dd < -APERT) v = -3.0e38f;
      Pt[hrow][il] = v;
    }
  }
  __syncthreads();

  // ---- phase 2: softmax over i for each column j --------------------------
  {
    const int j = tid & 15, sub = tid >> 4;    // 8 partials per column
    float m = -3.0e38f;
    for (int r = sub; r < ILEN; r += 8) m = fmaxf(m, Pt[j][r]);
    red[j][sub] = m;
    __syncthreads();
    float mm = red[j][0];
    #pragma unroll
    for (int s = 1; s < 8; ++s) mm = fmaxf(mm, red[j][s]);
    __syncthreads();
    float sum = 0.0f;
    for (int r = sub; r < ILEN; r += 8) {
      float e = expf(Pt[j][r] - mm);
      Pt[j][r] = e;
      sum += e;
    }
    red[j][sub] = sum;
    __syncthreads();
    float tot = 0.0f;
    #pragma unroll
    for (int s = 0; s < 8; ++s) tot += red[j][s];
    const float inv = 1.0f / tot;
    for (int r = sub; r < ILEN; r += 8) Pt[j][r] *= inv;
  }
  __syncthreads();

  // ---- phase 3: Y[j0:j0+16, :] = P^T @ V_band; wave owns 256 columns ------
  // K=32-deep reduction over the 544-row panel: 17 WMMA steps per n-chunk.
  v8f accs[16];
  #pragma unroll
  for (int c = 0; c < 16; ++c) accs[c] = (v8f){};

  for (int kt = 0; kt < ILEN / 32; ++kt) {
    const int kb = kt * 32;
    // A[m=j][k] = Pt[j][kb+k]; contiguous 8xf32 runs -> ds_load_b128 pairs.
    const float* pr0 = &Pt[hrow][kb + half * 8];
    const float* pr1 = pr0 + 16;
    v16bf a;
    #pragma unroll
    for (int t = 0; t < 8; ++t) {
      a[t]     = (__bf16)pr0[t];
      a[8 + t] = (__bf16)pr1[t];
    }
    // B[k][n] = V[ib+k][n] = Vt[n][ib+k]; two contiguous 8xbf16 runs.
    int b0 = i0 + kb + half * 8;
    int b1 = b0 + 16;
    b0 = (b0 < 0) ? 0 : (b0 > NSEQ - 8 ? NSEQ - 8 : b0);   // P==0 off-band
    b1 = (b1 < 0) ? 0 : (b1 > NSEQ - 8 ? NSEQ - 8 : b1);
    #pragma unroll
    for (int c = 0; c < 16; ++c) {
      const __bf16* vrow = Vt + (size_t)(wid * 256 + c * 16 + hrow) * NSEQ;
      v16bf b = frag_from_runs(vrow + b0, vrow + b1);
      accs[c] = wmma_bf16(a, b, accs[c]);
    }
  }

  #pragma unroll
  for (int c = 0; c < 16; ++c) {
    const int col = wid * 256 + c * 16 + hrow;
    #pragma unroll
    for (int r = 0; r < 8; ++r) {
      const int m = r + 8 * half;
      Y[(size_t)(j0 + m) * DMODEL + col] = accs[c][r];
    }
  }
}

// ---------------------------------------------------------------------------
// Row-wise LayerNorm in place: 1 block (256 threads) per row of 1024.
// ---------------------------------------------------------------------------
__global__ __launch_bounds__(256)
void layernorm_inplace(float* __restrict__ y, const float* __restrict__ g,
                       const float* __restrict__ b) {
  __shared__ float red[256];
  const int row = blockIdx.x;
  const int tid = threadIdx.x;
  float v[4];
  float s = 0.0f;
  #pragma unroll
  for (int i = 0; i < 4; ++i) {
    v[i] = y[(size_t)row * DMODEL + tid + i * 256];
    s += v[i];
  }
  red[tid] = s;
  __syncthreads();
  for (int off = 128; off; off >>= 1) {
    if (tid < off) red[tid] += red[tid + off];
    __syncthreads();
  }
  const float mu = red[0] * (1.0f / DMODEL);
  __syncthreads();
  float q = 0.0f;
  #pragma unroll
  for (int i = 0; i < 4; ++i) {
    float d = v[i] - mu;
    q += d * d;
  }
  red[tid] = q;
  __syncthreads();
  for (int off = 128; off; off >>= 1) {
    if (tid < off) red[tid] += red[tid + off];
    __syncthreads();
  }
  const float rstd = rsqrtf(red[0] * (1.0f / DMODEL) + LN_EPS);
  #pragma unroll
  for (int i = 0; i < 4; ++i) {
    const int c = tid + i * 256;
    y[(size_t)row * DMODEL + c] = (v[i] - mu) * rstd * g[c] + b[c];
  }
}

// ---------------------------------------------------------------------------
// Final GEMV + sigmoid: out[row] = sigmoid(dot(A[row,:], w) + b0)
// ---------------------------------------------------------------------------
__global__ __launch_bounds__(128)
void gemv_sigmoid(const float* __restrict__ A, const float* __restrict__ w,
                  const float* __restrict__ b, float* __restrict__ out) {
  __shared__ float red[128];
  const int row = blockIdx.x;
  const int tid = threadIdx.x;
  float s = 0.0f;
  for (int c = tid; c < DMODEL; c += 128)
    s += A[(size_t)row * DMODEL + c] * w[c];
  red[tid] = s;
  __syncthreads();
  for (int off = 64; off; off >>= 1) {
    if (tid < off) red[tid] += red[tid + off];
    __syncthreads();
  }
  if (tid == 0) out[row] = 1.0f / (1.0f + expf(-(red[0] + b[0])));
}

// ---------------------------------------------------------------------------
extern "C" void kernel_launch(void* const* d_in, const int* in_sizes, int n_in,
                              void* d_out, int out_size, void* d_ws, size_t ws_size,
                              hipStream_t stream) {
  const float* x    = (const float*)d_in[0];
  const float* wk   = (const float*)d_in[1];
  const float* wq   = (const float*)d_in[2];
  const float* wv   = (const float*)d_in[3];
  const float* wo   = (const float*)d_in[4];
  const float* wka  = (const float*)d_in[5];
  const float* bka  = (const float*)d_in[6];
  const float* wkd  = (const float*)d_in[7];
  const float* bkd  = (const float*)d_in[8];
  const float* g_y  = (const float*)d_in[9];
  const float* b_y  = (const float*)d_in[10];
  const float* g_ka = (const float*)d_in[11];
  const float* b_ka = (const float*)d_in[12];
  float* out = (float*)d_out;

  char* w = (char*)d_ws;
  __bf16* Kb   = (__bf16*)(w);
  __bf16* Qb   = (__bf16*)(w + (8ull  << 20));
  __bf16* Vt   = (__bf16*)(w + (16ull << 20));   // transposed V: [1024][4096]
  float*  yatt = (float*) (w + (24ull << 20));
  float*  y2   = (float*) (w + (40ull << 20));
  float*  t1   = (float*) (w + (56ull << 20));

  dim3 ggrid(NSEQ / BM, DMODEL / BN);   // 32 x 16

  // K, Q (pre-scaled 0.06) row-major; V transposed, all bf16.
  gemm_bf16out<false><<<ggrid, 256, 0, stream>>>(x, wk, Kb, 1.0f);
  gemm_bf16out<false><<<ggrid, 256, 0, stream>>>(x, wq, Qb, 0.06f);
  gemm_bf16out<true ><<<ggrid, 256, 0, stream>>>(x, wv, Vt, 1.0f);

  // banded attention, softmax over queries
  attn_banded<<<NSEQ / JT, 128, 0, stream>>>(Qb, Kb, Vt, yatt);

  // y2 = yatt @ wo + x ; LN
  gemm_f32out<0><<<ggrid, 256, 0, stream>>>(yatt, wo, y2, x, nullptr);
  layernorm_inplace<<<NSEQ, 256, 0, stream>>>(y2, g_y, b_y);

  // t1 = relu(y2 @ wka + bka) ; LN
  gemm_f32out<1><<<ggrid, 256, 0, stream>>>(y2, wka, t1, nullptr, bka);
  layernorm_inplace<<<NSEQ, 256, 0, stream>>>(t1, g_ka, b_ka);

  // out = sigmoid(t1 @ wkd + bkd)
  gemv_sigmoid<<<NSEQ, 128, 0, stream>>>(t1, wkd, bkd, out);
}
